// MotherCubeConv_75823352643952
// MI455X (gfx1250) — matrix operand order repaired and checked by reading, not compiled, not run on hardware
//
#include <hip/hip_runtime.h>

typedef __attribute__((ext_vector_type(16))) __bf16 v16bf;
typedef __attribute__((ext_vector_type(8)))  __bf16 v8bf;
typedef __attribute__((ext_vector_type(8)))  float  v8f;

#define IN_F   128
#define OUT_F  128
#define FANIN  640          // 5 * 128
#define MTILE  64           // tets per workgroup (4 x 16 per wave)
#define KSTEP  32           // bf16 WMMA K
#define NACC   (MTILE / 16) // accumulators per wave

// ---------------- fp32 -> bf16 (round-to-nearest-even) ----------------
static __device__ __forceinline__ unsigned int pack_bf2(float lo, float hi) {
    unsigned ul = __float_as_uint(lo);
    unsigned uh = __float_as_uint(hi);
    ul = (ul + (0x7FFFu + ((ul >> 16) & 1u))) >> 16;
    uh = (uh + (0x7FFFu + ((uh >> 16) & 1u))) >> 16;
    return (uh << 16) | (ul & 0xFFFFu);
}

__global__ void cvt_f32_to_bf16(const float* __restrict__ src,
                                unsigned short* __restrict__ dst,
                                long n) {
    long tid    = (long)blockIdx.x * blockDim.x + threadIdx.x;
    long stride = (long)gridDim.x * blockDim.x;
    // 8 floats -> 8 bf16 (16B store) per iteration; n is a multiple of 8.
    for (long e = tid * 8; e < n; e += stride * 8) {
        float4 a = *(const float4*)(src + e);
        float4 b = *(const float4*)(src + e + 4);
        uint4 o;
        o.x = pack_bf2(a.x, a.y);
        o.y = pack_bf2(a.z, a.w);
        o.z = pack_bf2(b.x, b.y);
        o.w = pack_bf2(b.z, b.w);
        *(uint4*)(dst + e) = o;
    }
}

static __device__ __forceinline__ v16bf cat16(v8bf lo, v8bf hi) {
    return __builtin_shufflevector(lo, hi, 0,1,2,3,4,5,6,7,8,9,10,11,12,13,14,15);
}

// ------------- gather + GEMM: out[64,128] = relu(A[64,640] x W^T + b) -------------
// One 256-thread block (8 wave32) per 64-tet tile. A staged in LDS (gathered),
// B (bf16 W, row-major [128][640]) read directly from global: per-lane
// k-contiguous data uses the same k order as the A fragments (lanes 0-15 hold
// K=0-15, lanes 16-31 hold K=16-31 of each 32-K step). Each wave owns a
// 16-wide N slice and amortizes every B fragment over 4 WMMAs (M=64).
__global__ void __launch_bounds__(256)
mothercube_wmma(const unsigned short* __restrict__ Fb,   // bf16 features [N][128]
                const unsigned short* __restrict__ Wb,   // bf16 W [128][640]
                const int*            __restrict__ nbr,  // [N][4]
                const float*          __restrict__ bias, // [128]
                float*                __restrict__ out,  // [N][128]
                int nTets) {
    __shared__ unsigned short sA[MTILE * FANIN];         // 80 KB

    const int tid   = threadIdx.x;
    const int block = blockIdx.x;
    const int tet0  = block * MTILE;

    // -------- stage A tile: 64 tets x 5 sources x 128 bf16, 16B per thread-op ----
    {
        const int g  = tid >> 4;       // 16 groups of 16 threads
        const int sl = tid & 15;       // 16B chunk within a 256B row-segment
        for (int seg = g; seg < MTILE * 5; seg += 16) {
            const int tl = seg / 5;            // local tet row
            const int s  = seg % 5;            // 0=self, 1..4=neighbors
            int tetG = tet0 + tl;
            if (tetG >= nTets) tetG = nTets - 1;
            int srcRow = (s == 0) ? tetG : nbr[(size_t)tetG * 4 + (s - 1)];
            const v8bf* src = (const v8bf*)(Fb + (size_t)srcRow * IN_F + sl * 8);
            *(v8bf*)(sA + tl * FANIN + s * IN_F + sl * 8) = *src;
        }
    }
    __syncthreads();

    // -------- per-wave 64(M) x 16(N) GEMM over K=640 --------
    const int wave = tid >> 5;         // 0..7 -> N tile
    const int lane = tid & 31;
    const int ln   = lane & 15;
    const int half = lane >> 4;

    const int nG = wave * 16 + ln;     // global output feature this lane covers

    v8f acc[NACC];
    #pragma unroll
    for (int a = 0; a < NACC; ++a) acc[a] = (v8f){};

    const unsigned short* wrow = Wb + (size_t)nG * FANIN;   // B: k-contiguous per lane

    #pragma unroll 2
    for (int kk = 0; kk < FANIN; kk += KSTEP) {
        // B fragment: 16 contiguous K values at kk + half*16
        const int kb = kk + half * 16;
        v16bf bfrag = cat16(*(const v8bf*)(wrow + kb),
                            *(const v8bf*)(wrow + kb + 8));

        // A fragments from LDS: per lane 16 contiguous K values at kk + half*16
        const int ka = kk + half * 16;
        #pragma unroll
        for (int a = 0; a < NACC; ++a) {
            const int row = a * 16 + ln;
            v16bf af = cat16(*(const v8bf*)(sA + row * FANIN + ka),
                             *(const v8bf*)(sA + row * FANIN + ka + 8));
            acc[a] = __builtin_amdgcn_wmma_f32_16x16x32_bf16(false, af, false, bfrag,
                                                             (short)0, acc[a],
                                                             false, false);
        }
    }

    // -------- epilogue: bias + ReLU, f32 store (branchless when tile is full) ----
    const float bn = bias[nG];
    float* obase = out + (size_t)tet0 * OUT_F + nG;
    if (tet0 + MTILE <= nTets) {
        #pragma unroll
        for (int a = 0; a < NACC; ++a) {
            #pragma unroll
            for (int r = 0; r < 8; ++r) {
                const int m = a * 16 + half * 8 + r;  // C/D: VGPR r -> M = r + 8*half
                obase[(size_t)m * OUT_F] = fmaxf(acc[a][r] + bn, 0.0f);
            }
        }
    } else {
        #pragma unroll
        for (int a = 0; a < NACC; ++a) {
            #pragma unroll
            for (int r = 0; r < 8; ++r) {
                const int m = a * 16 + half * 8 + r;
                if (tet0 + m < nTets)
                    obase[(size_t)m * OUT_F] = fmaxf(acc[a][r] + bn, 0.0f);
            }
        }
    }
}

extern "C" void kernel_launch(void* const* d_in, const int* in_sizes, int n_in,
                              void* d_out, int out_size, void* d_ws, size_t ws_size,
                              hipStream_t stream) {
    const float* features  = (const float*)d_in[0];  // [N,128] f32
    const int*   neighbors = (const int*)d_in[1];    // [N,4]
    const float* W         = (const float*)d_in[2];  // [128,640] f32
    const float* b         = (const float*)d_in[3];  // [128]
    float*       out       = (float*)d_out;

    const long featElems = in_sizes[0];              // N*128
    const long wElems    = in_sizes[2];              // 128*640
    const int  nTets     = (int)(featElems / IN_F);

    // Workspace layout: [features bf16][W bf16]
    unsigned short* Fb = (unsigned short*)d_ws;
    unsigned short* Wb = Fb + featElems;             // featElems*2 bytes is 256B-aligned

    // 1) convert features and W to bf16
    {
        int blk = 256;
        int grd = (int)((featElems / 8 + blk - 1) / blk);
        if (grd > 4096) grd = 4096;
        cvt_f32_to_bf16<<<grd, blk, 0, stream>>>(features, Fb, featElems);
        int grdW = (int)((wElems / 8 + blk - 1) / blk);
        cvt_f32_to_bf16<<<grdW, blk, 0, stream>>>(W, Wb, wElems);
    }

    // 2) gather + WMMA GEMM + bias + ReLU
    {
        int grd = (nTets + MTILE - 1) / MTILE;       // 3125 blocks for N=200000
        mothercube_wmma<<<grd, 256, 0, stream>>>(Fb, Wb, neighbors, b, out, nTets);
    }
    (void)n_in; (void)out_size; (void)ws_size;
}
